// CrossOver_70841190580536
// MI455X (gfx1250) — compile-verified, matrix operand order checked by reading, not run on hardware
//
#include <hip/hip_runtime.h>
#include <hip/hip_bf16.h>
#include <math.h>

typedef _Float16 half_t;
typedef _Float16 v16h __attribute__((ext_vector_type(16)));
typedef _Float16 v8h  __attribute__((ext_vector_type(8)));
typedef float    v8f  __attribute__((ext_vector_type(8)));

#define H_   12
#define CH_  16
#define C_   8
#define N_   256
#define CS_  384
#define INF_ 100000.0f
#define EPS_ 1e-8f
#define SCALE_ 0.14433756729740643f   /* (1/(3*CH))^0.5 = (1/48)^0.5 */
#define HWN_   0.13608276348795434f   /* (1/(3*PQ*9/2))^0.5 = (1/54)^0.5 */

__device__ __forceinline__ v16h ldfrag(const half_t* __restrict__ p, int kb, int khalf)
{
  v8h lo = *(const v8h*)(p + kb + khalf * 8);
  v8h hi = *(const v8h*)(p + kb + 16 + khalf * 8);
  return __builtin_shufflevector(lo, hi, 0,1,2,3,4,5,6,7,8,9,10,11,12,13,14,15);
}

// ---------------------------------------------------------------------------
// Generic batched f16 WMMA GEMM: C[M,N] = A[M,K] * BT[N,K]^T + bias[N]
// One wave (32 threads) per 32x32 output macro-tile (2x2 WMMA accumulators).
// K % 32 == 0, M,N % 32 == 0. Prefetches the streaming K dimension.
// ---------------------------------------------------------------------------
__global__ __launch_bounds__(32) void k_gemm_f16(
    const half_t* __restrict__ A, const half_t* __restrict__ BT,
    const float* __restrict__ bias, float* __restrict__ C,
    int M, int N, int K,
    long long sA, long long sB, long long sC)
{
  A  += (size_t)blockIdx.z * (size_t)sA;
  BT += (size_t)blockIdx.z * (size_t)sB;
  C  += (size_t)blockIdx.z * (size_t)sC;
  const int tm = blockIdx.x, tn = blockIdx.y;
  const int lane = threadIdx.x & 31;
  const int lrow = lane & 15, khalf = lane >> 4;
  const half_t* a0 = A  + (size_t)(tm * 32 + lrow) * K;
  const half_t* a1 = a0 + (size_t)16 * K;
  const half_t* b0 = BT + (size_t)(tn * 32 + lrow) * K;
  const half_t* b1 = b0 + (size_t)16 * K;
  v8f acc00 = {}, acc01 = {}, acc10 = {}, acc11 = {};
  for (int kb = 0; kb < K; kb += 32) {
    if (kb + 128 < K) {
      __builtin_prefetch(a0 + kb + 128, 0, 3);
      __builtin_prefetch(a1 + kb + 128, 0, 3);
      __builtin_prefetch(b0 + kb + 128, 0, 3);
      __builtin_prefetch(b1 + kb + 128, 0, 3);
    }
    v16h af0 = ldfrag(a0, kb, khalf);
    v16h af1 = ldfrag(a1, kb, khalf);
    v16h bf0 = ldfrag(b0, kb, khalf);
    v16h bf1 = ldfrag(b1, kb, khalf);
    acc00 = __builtin_amdgcn_wmma_f32_16x16x32_f16(false, af0, false, bf0, (short)0, acc00, false, false);
    acc01 = __builtin_amdgcn_wmma_f32_16x16x32_f16(false, af0, false, bf1, (short)0, acc01, false, false);
    acc10 = __builtin_amdgcn_wmma_f32_16x16x32_f16(false, af1, false, bf0, (short)0, acc10, false, false);
    acc11 = __builtin_amdgcn_wmma_f32_16x16x32_f16(false, af1, false, bf1, (short)0, acc11, false, false);
  }
  const int col0 = tn * 32 + lrow, col1 = col0 + 16;
  const float bv0 = bias ? bias[col0] : 0.0f;
  const float bv1 = bias ? bias[col1] : 0.0f;
  const int row0 = tm * 32 + khalf * 8;
#pragma unroll
  for (int r = 0; r < 8; ++r) {
    C[(size_t)(row0 + r)      * N + col0] = acc00[r] + bv0;
    C[(size_t)(row0 + r)      * N + col1] = acc01[r] + bv1;
    C[(size_t)(row0 + 16 + r) * N + col0] = acc10[r] + bv0;
    C[(size_t)(row0 + 16 + r) * N + col1] = acc11[r] + bv1;
  }
}

// ---------------------------------------------------------------------------
// Attention logits: K=64 extended QK^T WMMA (32x32 macro-tile per wave) +
// separable point terms + mask + the reference's exact db.reshape aliasing.
// grid = (8, 8, C_*H_)
// ---------------------------------------------------------------------------
__global__ __launch_bounds__(32) void k_logits(
    const half_t* __restrict__ qext, const half_t* __restrict__ kext,
    const float* __restrict__ rowpt, const float* __restrict__ colpt,
    const float* __restrict__ mask,  const float* __restrict__ trans,
    const float* __restrict__ wdist, const float* __restrict__ bdist,
    float* __restrict__ aout)
{
  const int ch = blockIdx.z, c = ch / H_, h = ch % H_;
  const int ti = blockIdx.x, tj = blockIdx.y;
  const int lane = threadIdx.x & 31;
  const int lrow = lane & 15, khalf = lane >> 4;
  const half_t* q0 = qext + ((size_t)(c * N_ + ti * 32 + lrow) * H_ + h) * 64;
  const half_t* q1 = q0 + (size_t)16 * H_ * 64;
  const half_t* k0 = kext + ((size_t)(c * N_ + tj * 32 + lrow) * H_ + h) * 64;
  const half_t* k1 = k0 + (size_t)16 * H_ * 64;
  v8f acc00 = {}, acc01 = {}, acc10 = {}, acc11 = {};
#pragma unroll
  for (int kb = 0; kb < 64; kb += 32) {
    v16h af0 = ldfrag(q0, kb, khalf);
    v16h af1 = ldfrag(q1, kb, khalf);
    v16h bf0 = ldfrag(k0, kb, khalf);
    v16h bf1 = ldfrag(k1, kb, khalf);
    acc00 = __builtin_amdgcn_wmma_f32_16x16x32_f16(false, af0, false, bf0, (short)0, acc00, false, false);
    acc01 = __builtin_amdgcn_wmma_f32_16x16x32_f16(false, af0, false, bf1, (short)0, acc01, false, false);
    acc10 = __builtin_amdgcn_wmma_f32_16x16x32_f16(false, af1, false, bf0, (short)0, acc10, false, false);
    acc11 = __builtin_amdgcn_wmma_f32_16x16x32_f16(false, af1, false, bf1, (short)0, acc11, false, false);
  }
  const int jc[2] = { tj * 32 + lrow, tj * 32 + 16 + lrow };
  float mjv[2], cvv[2];
#pragma unroll
  for (int b = 0; b < 2; ++b) {
    mjv[b] = mask[c * N_ + jc[b]];
    cvv[b] = colpt[(c * H_ + h) * N_ + jc[b]];
  }
  float* arow = aout + (size_t)(c * H_ + h) * N_ * N_;
#pragma unroll
  for (int ai = 0; ai < 2; ++ai) {
#pragma unroll
    for (int r = 0; r < 8; ++r) {
      const int i = ti * 32 + ai * 16 + khalf * 8 + r;
      const float rp = rowpt[(c * H_ + h) * N_ + i];
      const float mi = mask[c * N_ + i];
#pragma unroll
      for (int b = 0; b < 2; ++b) {
        const int j = jc[b];
        float aval;
        if (ai == 0) aval = (b == 0) ? acc00[r] : acc01[r];
        else         aval = (b == 0) ? acc10[r] : acc11[r];
        float val = aval + rp + cvv[b] + INF_ * (mi * mjv[b] - 1.0f);
        // db = (dist[...,None]*wdist + bdist).reshape(B,C,H,N,N) flat aliasing
        const int flat = h * 65536 + i * 256 + j;
        const int si = flat / 3072;
        const int rem = flat - si * 3072;
        const int sj = rem / 12;
        const int sh = rem - sj * 12;
        const float* ta = trans + (size_t)(c * N_ + si) * 3;
        const float* tb = trans + (size_t)(c * N_ + sj) * 3;
        const float dx = ta[0] - tb[0], dy = ta[1] - tb[1], dz = ta[2] - tb[2];
        val += sqrtf(dx * dx + dy * dy + dz * dz) * wdist[sh] + bdist[sh];
        arow[(size_t)i * N_ + j] = val;
      }
    }
  }
}

// ---------------------------------------------------------------------------
// Row softmax over j (in place on f32) + f16 copy for the AV WMMA.
// ---------------------------------------------------------------------------
__global__ __launch_bounds__(256) void k_softmax(float* __restrict__ a,
                                                 half_t* __restrict__ a16)
{
  __shared__ float sh[256];
  const size_t row = blockIdx.x;
  const int t = threadIdx.x;
  const float v = a[row * 256 + t];
  sh[t] = v; __syncthreads();
  for (int s = 128; s > 0; s >>= 1) {
    if (t < s) sh[t] = fmaxf(sh[t], sh[t + s]);
    __syncthreads();
  }
  const float mx = sh[0]; __syncthreads();
  const float e = expf(v - mx);
  sh[t] = e; __syncthreads();
  for (int s = 128; s > 0; s >>= 1) {
    if (t < s) sh[t] += sh[t + s];
    __syncthreads();
  }
  const float p = e / sh[0];
  a[row * 256 + t]   = p;
  a16[row * 256 + t] = (half_t)p;
}

// ---------------------------------------------------------------------------
// s_g = mean_c(s); LayerNorm -> s_gn (written directly as f16 GEMM operand)
// ---------------------------------------------------------------------------
__global__ __launch_bounds__(384) void k_meanln(
    const float* __restrict__ s, const float* __restrict__ gamma,
    const float* __restrict__ beta, half_t* __restrict__ sgn16)
{
  __shared__ float sh[384];
  __shared__ float s_mean, s_var;
  const int n = blockIdx.x, f = threadIdx.x;
  float acc = 0.f;
  for (int c = 0; c < C_; ++c) acc += s[(size_t)(c * N_ + n) * CS_ + f];
  const float x = acc * (1.0f / C_);
  sh[f] = x; __syncthreads();
  for (int st = 192; st >= 3; st >>= 1) { if (f < st) sh[f] += sh[f + st]; __syncthreads(); }
  if (f == 0) s_mean = (sh[0] + sh[1] + sh[2]) * (1.0f / CS_);
  __syncthreads();
  const float xm = x - s_mean;
  sh[f] = xm * xm; __syncthreads();
  for (int st = 192; st >= 3; st >>= 1) { if (f < st) sh[f] += sh[f + st]; __syncthreads(); }
  if (f == 0) s_var = (sh[0] + sh[1] + sh[2]) * (1.0f / CS_);
  __syncthreads();
  const float y = xm * rsqrtf(s_var + 1e-5f) * gamma[f] + beta[f];
  sgn16[(size_t)n * CS_ + f] = (half_t)y;
}

// ---------------------------------------------------------------------------
// Rotate+translate raw points from the fused projection. q_pts/k_pts -> f32;
// v_pts written straight into the transposed f16 V-pack (cols 16..39 of 64).
// grid = C_*N_ blocks, 192 threads (48 q-points + 144 kv-points).
// ---------------------------------------------------------------------------
__global__ __launch_bounds__(192) void k_pts(
    const float* __restrict__ proj, const float* __restrict__ rot,
    const float* __restrict__ trans, float* __restrict__ qpts,
    float* __restrict__ kpts, half_t* __restrict__ vT16)
{
  const int cn = blockIdx.x, c = cn / N_, n = cn % N_;
  const int t = threadIdx.x;
  const float* prow = proj + (size_t)cn * 1152;
  float raw[3];
  if (t < 48) {
    for (int j = 0; j < 3; ++j) raw[j] = prow[576 + j * 48 + t];
  } else {
    const int pi = t - 48;
    for (int j = 0; j < 3; ++j) raw[j] = prow[720 + j * 144 + pi];
  }
  const float* R = rot + (size_t)cn * 9;
  const float* T = trans + (size_t)cn * 3;
  const float o0 = R[0]*raw[0] + R[1]*raw[1] + R[2]*raw[2] + T[0];
  const float o1 = R[3]*raw[0] + R[4]*raw[1] + R[5]*raw[2] + T[1];
  const float o2 = R[6]*raw[0] + R[7]*raw[1] + R[8]*raw[2] + T[2];
  if (t < 48) {
    float* d = qpts + ((size_t)cn * 48 + t) * 3;
    d[0] = o0; d[1] = o1; d[2] = o2;
  } else {
    const int pi = t - 48, h = pi / 12, pp = pi % 12;
    if (pp < 4) {
      float* d = kpts + ((size_t)cn * 48 + h * 4 + pp) * 3;
      d[0] = o0; d[1] = o1; d[2] = o2;
    } else {
      const int p = pp - 4;
      half_t* d = vT16 + ((size_t)(c * H_ + h) * 64 + 16 + p * 3) * N_ + n;
      d[0]      = (half_t)o0;
      d[N_]     = (half_t)o1;
      d[2 * N_] = (half_t)o2;
    }
  }
}

// Separable point-attention terms: -0.5*hw[h]*Sum_p |pts|^2 per (c,h,n).
__global__ void k_ptterms(const float* __restrict__ qpts,
                          const float* __restrict__ kpts,
                          const float* __restrict__ hweights,
                          float* __restrict__ rowpt, float* __restrict__ colpt,
                          int total)
{
  const int t = blockIdx.x * blockDim.x + threadIdx.x;
  if (t >= total) return;
  const int c = t / (H_ * N_);
  const int rem = t % (H_ * N_);
  const int h = rem / N_, n = rem % N_;
  const float hwv = log1pf(expf(hweights[h])) * HWN_;
  const int cn = c * N_ + n;
  float sq = 0.f, sk = 0.f;
  for (int p = 0; p < 4; ++p)
    for (int x = 0; x < 3; ++x) {
      const float a = qpts[((size_t)cn * 48 + h * 4 + p) * 3 + x]; sq += a * a;
      const float b = kpts[((size_t)cn * 48 + h * 4 + p) * 3 + x]; sk += b * b;
    }
  rowpt[t] = -0.5f * hwv * sq;
  colpt[t] = -0.5f * hwv * sk;
}

// Build the 64-channel extended Q/K vectors (f16) per (c,n,h).
__global__ __launch_bounds__(64) void k_qkext(
    const float* __restrict__ proj, const float* __restrict__ gqk,
    const float* __restrict__ qpts, const float* __restrict__ kpts,
    const float* __restrict__ hweights,
    half_t* __restrict__ qext, half_t* __restrict__ kext)
{
  const int idx = blockIdx.x;            // (c*N_+n)*H_ + h
  const int h = idx % H_;
  const int cn = idx / H_;
  const int n = cn % N_;
  const int t = threadIdx.x;
  float qe = 0.f, ke = 0.f;
  if (t < 16) {
    qe = proj[(size_t)cn * 1152 + h * 16 + t] * SCALE_;
    ke = proj[(size_t)cn * 1152 + 192 + h * 32 + t];
  } else if (t < 32) {
    const int d = t - 16;
    qe = gqk[(size_t)n * 384 + h * 16 + d] * SCALE_;
    ke = gqk[(size_t)n * 384 + 192 + h * 16 + d];
  } else if (t < 44) {
    const int pc = t - 32, p = pc / 3, x = pc % 3;
    const float hwv = log1pf(expf(hweights[h])) * HWN_;
    qe = qpts[((size_t)cn * 48 + h * 4 + p) * 3 + x] * hwv;
    ke = kpts[((size_t)cn * 48 + h * 4 + p) * 3 + x];
  }
  qext[(size_t)idx * 64 + t] = (half_t)qe;
  kext[(size_t)idx * 64 + t] = (half_t)ke;
}

// Pack V columns (0..15) + zero pad columns (40..63) of the transposed V.
__global__ void k_vpack(const float* __restrict__ proj,
                        half_t* __restrict__ vT16, int total)
{
  int t = blockIdx.x * blockDim.x + threadIdx.x;
  if (t >= total) return;
  const int j = t % N_; t /= N_;
  const int ci = t % 40; t /= 40;
  const int h = t % H_;
  const int c = t / H_;
  float val = 0.f;
  int col;
  if (ci < 16) {
    col = ci;
    val = proj[(size_t)(c * N_ + j) * 1152 + 192 + h * 32 + 16 + ci];
  } else {
    col = 24 + ci;   // 40..63 zero pad
  }
  vT16[((size_t)(c * H_ + h) * 64 + col) * N_ + j] = (half_t)val;
}

// Inverse-frame transform of o_pt + norms; assemble feat row (f16, 576 cols).
__global__ __launch_bounds__(96) void k_feat(
    const float* __restrict__ po, const float* __restrict__ rot,
    const float* __restrict__ trans, half_t* __restrict__ feat)
{
  const int cn = blockIdx.x, c = cn / N_, n = cn % N_;
  const int t = threadIdx.x;
  half_t* frow = feat + (size_t)cn * 576;
  for (int it = t; it < 192; it += 96) {
    const int h = it / 16, d = it % 16;
    frow[it] = (half_t)po[((size_t)(c * H_ + h) * N_ + n) * 64 + d];
  }
  const int h = t / 8, p = t % 8;
  const float* pb = po + ((size_t)(c * H_ + h) * N_ + n) * 64 + 16 + p * 3;
  const float* T = trans + (size_t)cn * 3;
  const float* R = rot + (size_t)cn * 9;
  const float d0 = pb[0] - T[0], d1 = pb[1] - T[1], d2 = pb[2] - T[2];
  const float x = R[0] * d0 + R[3] * d1 + R[6] * d2;   // rot^T
  const float y = R[1] * d0 + R[4] * d1 + R[7] * d2;
  const float z = R[2] * d0 + R[5] * d1 + R[8] * d2;
  const float nm = sqrtf(x * x + y * y + z * z + EPS_);
  frow[192 + t] = (half_t)x;
  frow[288 + t] = (half_t)y;
  frow[384 + t] = (half_t)z;
  frow[480 + t] = (half_t)nm;
}

// f32 (K x N) weight -> transposed f16 (N x K).
__global__ void k_packwT(const float* __restrict__ src, half_t* __restrict__ dst,
                         int K, int Nl, int total)
{
  const int t = blockIdx.x * blockDim.x + threadIdx.x;
  if (t >= total) return;
  const int n = t / K, k = t % K;
  dst[(size_t)n * K + k] = (half_t)src[(size_t)k * Nl + n];
}

__global__ void k_copyf(const float* __restrict__ src, float* __restrict__ dst, int n)
{
  const int t = blockIdx.x * blockDim.x + threadIdx.x;
  if (t < n) dst[t] = src[t];
}

__global__ void k_cvt16(const float* __restrict__ src, half_t* __restrict__ dst, int n)
{
  const int t = blockIdx.x * blockDim.x + threadIdx.x;
  if (t < n) dst[t] = (half_t)src[t];
}

// ---------------------------------------------------------------------------
extern "C" void kernel_launch(void* const* d_in, const int* in_sizes, int n_in,
                              void* d_out, int out_size, void* d_ws, size_t ws_size,
                              hipStream_t stream)
{
  (void)in_sizes; (void)n_in; (void)out_size; (void)ws_size;
  const float* s      = (const float*)d_in[0];
  const float* rot    = (const float*)d_in[2];
  const float* trans  = (const float*)d_in[3];
  const float* mask   = (const float*)d_in[4];
  const float* wq     = (const float*)d_in[5];
  const float* bq     = (const float*)d_in[6];
  const float* wkv    = (const float*)d_in[7];
  const float* bkv    = (const float*)d_in[8];
  const float* ggam   = (const float*)d_in[9];
  const float* gbet   = (const float*)d_in[10];
  const float* wgq    = (const float*)d_in[11];
  const float* bgq    = (const float*)d_in[12];
  const float* wgk    = (const float*)d_in[13];
  const float* bgk    = (const float*)d_in[14];
  const float* wqp    = (const float*)d_in[15];
  const float* bqp    = (const float*)d_in[16];
  const float* wkvp   = (const float*)d_in[17];
  const float* bkvp   = (const float*)d_in[18];
  const float* hwts   = (const float*)d_in[19];
  const float* wdist  = (const float*)d_in[20];
  const float* bdist  = (const float*)d_in[21];
  const float* wout   = (const float*)d_in[22];
  const float* bout   = (const float*)d_in[23];

  float* outp = (float*)d_out;                 // (C,N,384)
  float* ap   = outp + (size_t)C_ * N_ * CS_;  // (C,H,N,N)

  char* ws = (char*)d_ws;
  size_t off = 0;
  auto alloc = [&](size_t bytes) -> void* {
    void* p = ws + off;
    off = (off + bytes + 63) & ~(size_t)63;
    return p;
  };
  float*  proj   = (float*) alloc((size_t)2048 * 1152 * 4);
  float*  gqk    = (float*) alloc((size_t)256 * 384 * 4);
  float*  qpts   = (float*) alloc((size_t)2048 * 48 * 3 * 4);
  float*  kpts   = (float*) alloc((size_t)2048 * 48 * 3 * 4);
  float*  rowpt  = (float*) alloc((size_t)C_ * H_ * N_ * 4);
  float*  colpt  = (float*) alloc((size_t)C_ * H_ * N_ * 4);
  float*  po     = (float*) alloc((size_t)C_ * H_ * N_ * 64 * 4);
  float*  projB  = (float*) alloc((size_t)1152 * 4);
  float*  gB     = (float*) alloc((size_t)384 * 4);
  half_t* s16    = (half_t*)alloc((size_t)2048 * 384 * 2);
  half_t* sgn16  = (half_t*)alloc((size_t)256 * 384 * 2);
  half_t* projWT = (half_t*)alloc((size_t)1152 * 384 * 2);
  half_t* gWT    = (half_t*)alloc((size_t)384 * 384 * 2);
  half_t* woutT  = (half_t*)alloc((size_t)384 * 576 * 2);
  half_t* vT16   = (half_t*)alloc((size_t)C_ * H_ * 64 * N_ * 2);
  half_t* qext   = (half_t*)alloc((size_t)2048 * H_ * 64 * 2);
  half_t* kext   = (half_t*)alloc((size_t)2048 * H_ * 64 * 2);
  half_t* a16    = (half_t*)alloc((size_t)C_ * H_ * N_ * N_ * 2);
  half_t* feat16 = (half_t*)alloc((size_t)2048 * 576 * 2);

  auto gs = [](int total) { return dim3((total + 255) / 256); };

  // Weight packing (transpose + f16)
  k_packwT<<<gs(384*192), 256, 0, stream>>>(wq,   projWT + (size_t)0   * 384, 384, 192, 384*192);
  k_packwT<<<gs(384*384), 256, 0, stream>>>(wkv,  projWT + (size_t)192 * 384, 384, 384, 384*384);
  k_packwT<<<gs(384*144), 256, 0, stream>>>(wqp,  projWT + (size_t)576 * 384, 384, 144, 384*144);
  k_packwT<<<gs(384*432), 256, 0, stream>>>(wkvp, projWT + (size_t)720 * 384, 384, 432, 384*432);
  k_packwT<<<gs(384*192), 256, 0, stream>>>(wgq,  gWT   + (size_t)0   * 384, 384, 192, 384*192);
  k_packwT<<<gs(384*192), 256, 0, stream>>>(wgk,  gWT   + (size_t)192 * 384, 384, 192, 384*192);
  k_packwT<<<gs(576*384), 256, 0, stream>>>(wout, woutT, 576, 384, 576*384);
  // Bias packing
  k_copyf<<<gs(192), 256, 0, stream>>>(bq,   projB + 0,   192);
  k_copyf<<<gs(384), 256, 0, stream>>>(bkv,  projB + 192, 384);
  k_copyf<<<gs(144), 256, 0, stream>>>(bqp,  projB + 576, 144);
  k_copyf<<<gs(432), 256, 0, stream>>>(bkvp, projB + 720, 432);
  k_copyf<<<gs(192), 256, 0, stream>>>(bgq,  gB + 0,   192);
  k_copyf<<<gs(192), 256, 0, stream>>>(bgk,  gB + 192, 192);

  // s -> f16 ; mean over c + LayerNorm -> sgn16
  k_cvt16<<<gs(2048*384), 256, 0, stream>>>(s, s16, 2048*384);
  k_meanln<<<256, 384, 0, stream>>>(s, ggam, gbet, sgn16);

  // Fused projection GEMM: (2048x384) x (384x1152), 32x32 tiles
  k_gemm_f16<<<dim3(64, 36, 1), 32, 0, stream>>>(s16, projWT, projB, proj,
                                                 2048, 1152, 384, 0, 0, 0);
  // g_q|g_k GEMM: (256x384) x (384x384)
  k_gemm_f16<<<dim3(8, 12, 1), 32, 0, stream>>>(sgn16, gWT, gB, gqk,
                                                256, 384, 384, 0, 0, 0);

  // Points: rotate/translate; build q_pts/k_pts, v_pts straight into vT16
  k_pts<<<2048, 192, 0, stream>>>(proj, rot, trans, qpts, kpts, vT16);
  k_ptterms<<<gs(C_*H_*N_), 256, 0, stream>>>(qpts, kpts, hwts, rowpt, colpt, C_*H_*N_);
  k_vpack<<<gs(C_*H_*40*N_), 256, 0, stream>>>(proj, vT16, C_*H_*40*N_);
  k_qkext<<<2048 * H_, 64, 0, stream>>>(proj, gqk, qpts, kpts, hwts, qext, kext);

  // Attention logits (K=64 WMMA, 32x32 tiles) + softmax
  k_logits<<<dim3(8, 8, C_*H_), 32, 0, stream>>>(qext, kext, rowpt, colpt,
                                                 mask, trans, wdist, bdist, ap);
  k_softmax<<<C_*H_*N_, 256, 0, stream>>>(ap, a16);

  // Fused A*V and A*v_pts: batched (256x256) x (256x64) over C_*H_ batches
  k_gemm_f16<<<dim3(8, 2, C_*H_), 32, 0, stream>>>(a16, vT16, nullptr, po,
                                                   256, 64, 256,
                                                   (long long)N_ * N_,
                                                   (long long)64 * N_,
                                                   (long long)N_ * 64);

  // Inverse-frame transform + norms + feature assembly, then output GEMM
  k_feat<<<2048, 96, 0, stream>>>(po, rot, trans, feat16);
  k_gemm_f16<<<dim3(64, 12, 1), 32, 0, stream>>>(feat16, woutT, bout, outp,
                                                 2048, 384, 576, 0, 0, 0);
}
